// GINELayer_72421738545669
// MI455X (gfx1250) — matrix operand order repaired and checked by reading, not compiled, not run on hardware
//
#include <hip/hip_runtime.h>
#include <hip/hip_bf16.h>

typedef __bf16 v16bf __attribute__((ext_vector_type(16)));
typedef float  v8f   __attribute__((ext_vector_type(8)));

#define CCH 128   // channels

// K index base inside a 32-wide K tile for 16-bit A-matrix fragment layout:
// VGPR v (v<4): K=2v,2v+1 (low lanes) / +8 (high lanes); v>=4: K=16+2(v-4) / +8.
__device__ __forceinline__ int a_kbase(int v, int hi) {
    return (v < 4 ? 2 * v : 16 + 2 * (v - 4)) + (hi ? 8 : 0);
}

// ---------------------------------------------------------------- zero scratch
__global__ void gine_zero(float* p, int n) {
    int i = blockIdx.x * blockDim.x + threadIdx.x;
    if (i < n) p[i] = 0.0f;
}

// ------------------------------------------------- pack weights into B-fragment
// blob element index r = ((nt*4+kt)*32 + lane)*16 + i
// B 32x16 bf16 layout: lane l holds column n = l%16, K = (l<16?0:8)+i (i<8), 16+(l<16?0:8)+(i-8) (i>=8).
__global__ void gine_pack_w(const float* __restrict__ w1,
                            const float* __restrict__ w2,
                            const float* __restrict__ rw,
                            __bf16* __restrict__ blob) {
    int idx = blockIdx.x * blockDim.x + threadIdx.x;
    if (idx >= 3 * 16384) return;
    int w = idx / 16384, r = idx % 16384;
    int i  = r & 15;
    int l  = (r >> 4) & 31;
    int kt = (r >> 9) & 3;
    int nt = r >> 11;
    int k = kt * 32 + ((l >= 16) ? 8 : 0) + i + ((i >= 8) ? 8 : 0);
    int n = nt * 16 + (l & 15);
    const float* W = (w == 0) ? w1 : ((w == 1) ? w2 : rw);
    blob[idx] = (__bf16)W[k * CCH + n];
}

// ---------------------------------------------------------------- edge scatter
// One wave per edge; lane handles 4 channels. sums[dst] += x[src] + attr*ew + eb
__global__ void gine_edge(const float* __restrict__ x,
                          const long long* __restrict__ ei,
                          const float* __restrict__ eattr,
                          const float* __restrict__ ew,
                          const float* __restrict__ eb,
                          float* __restrict__ sums,
                          float* __restrict__ cnt, int E) {
    long long t = (long long)blockIdx.x * blockDim.x + threadIdx.x;
    int e    = (int)(t >> 5);
    int lane = (int)(t & 31);
    if (e >= E) return;
    long long s = ei[e];
    long long d = ei[(long long)E + e];
    float a = eattr[e];
    int c0 = lane * 4;
    float4 xv = *(const float4*)&x[(size_t)s * CCH + c0];
    float4 wv = *(const float4*)&ew[c0];
    float4 bv = *(const float4*)&eb[c0];
    float* dp = &sums[(size_t)d * CCH + c0];
    atomicAdd(dp + 0, xv.x + a * wv.x + bv.x);
    atomicAdd(dp + 1, xv.y + a * wv.y + bv.y);
    atomicAdd(dp + 2, xv.z + a * wv.z + bv.z);
    atomicAdd(dp + 3, xv.w + a * wv.w + bv.w);
    if (lane == 0) atomicAdd(&cnt[d], 1.0f);
}

// -------------------------------------------------------- fused node MLP (WMMA)
// One wave (32 threads) per 16-row tile. 96 v_wmma_f32_16x16x32_bf16 per wave.
__global__ void __launch_bounds__(32)
gine_node(const float* __restrict__ x,
          const float* __restrict__ sums,
          const float* __restrict__ cnt,
          const __bf16* __restrict__ blob,
          const float* __restrict__ b1,
          const float* __restrict__ b2,
          const float* __restrict__ rb,
          const float* __restrict__ epsp,
          float* __restrict__ out_pre,
          float* __restrict__ colsum,
          float* __restrict__ colsumsq, int N) {
    __shared__ __bf16 lds_t[16][CCH];

    const int lane = threadIdx.x;
    const int tile = blockIdx.x;
    const int m    = lane & 15;
    const int hi   = lane >> 4;
    const int row  = tile * 16 + m;
    const bool rowok = row < N;

    const float epsv = 1.0f + epsp[0];
    float inv = 1.0f;
    if (rowok) inv = 1.0f / fmaxf(cnt[row], 1.0f);

    // Build A fragments for h = (1+eps)*x + aggr and for x (residual GEMM).
    v16bf hfrag[4], xfrag[4];
#pragma unroll
    for (int kt = 0; kt < 4; ++kt) {
#pragma unroll
        for (int v = 0; v < 8; ++v) {
            int k = kt * 32 + a_kbase(v, hi);
            float2 xv = make_float2(0.0f, 0.0f), sv = make_float2(0.0f, 0.0f);
            if (rowok) {
                xv = *(const float2*)&x[(size_t)row * CCH + k];
                sv = *(const float2*)&sums[(size_t)row * CCH + k];
            }
            hfrag[kt][2 * v]     = (__bf16)(epsv * xv.x + sv.x * inv);
            hfrag[kt][2 * v + 1] = (__bf16)(epsv * xv.y + sv.y * inv);
            xfrag[kt][2 * v]     = (__bf16)xv.x;
            xfrag[kt][2 * v + 1] = (__bf16)xv.y;
        }
    }

    const __bf16* w1b = blob;
    const __bf16* w2b = blob + 16384;
    const __bf16* rwb = blob + 2 * 16384;

    // GEMM1: t = relu(h @ w1 + b1)  -> LDS (bf16)
#pragma unroll
    for (int nt = 0; nt < 8; ++nt) {
        int col = nt * 16 + m;
        float bb = b1[col];
        v8f acc = {bb, bb, bb, bb, bb, bb, bb, bb};
#pragma unroll
        for (int kt = 0; kt < 4; ++kt) {
            v16bf bf = *(const v16bf*)(w1b + ((size_t)(nt * 4 + kt) * 32 + lane) * 16);
            acc = __builtin_amdgcn_wmma_f32_16x16x32_bf16(
                false, hfrag[kt], false, bf, (short)0, acc, false, false);
        }
#pragma unroll
        for (int r = 0; r < 8; ++r)
            lds_t[r + hi * 8][col] = (__bf16)fmaxf(acc[r], 0.0f);
    }
    __syncthreads();

    // Reload t as A fragments.
    v16bf tfrag[4];
#pragma unroll
    for (int kt = 0; kt < 4; ++kt) {
#pragma unroll
        for (int v = 0; v < 8; ++v) {
            int k = kt * 32 + a_kbase(v, hi);
            tfrag[kt][2 * v]     = lds_t[m][k];
            tfrag[kt][2 * v + 1] = lds_t[m][k + 1];
        }
    }

    // GEMM2 + residual GEMM: out_pre = t @ w2 + b2 + x @ res_w + res_b
#pragma unroll
    for (int nt = 0; nt < 8; ++nt) {
        int col = nt * 16 + m;
        float bb = b2[col] + rb[col];
        v8f acc = {bb, bb, bb, bb, bb, bb, bb, bb};
#pragma unroll
        for (int kt = 0; kt < 4; ++kt) {
            v16bf bf = *(const v16bf*)(w2b + ((size_t)(nt * 4 + kt) * 32 + lane) * 16);
            acc = __builtin_amdgcn_wmma_f32_16x16x32_bf16(
                false, tfrag[kt], false, bf, (short)0, acc, false, false);
        }
#pragma unroll
        for (int kt = 0; kt < 4; ++kt) {
            v16bf bf = *(const v16bf*)(rwb + ((size_t)(nt * 4 + kt) * 32 + lane) * 16);
            acc = __builtin_amdgcn_wmma_f32_16x16x32_bf16(
                false, xfrag[kt], false, bf, (short)0, acc, false, false);
        }
        // store + column partial stats for BatchNorm
        float s = 0.0f, s2 = 0.0f;
#pragma unroll
        for (int r = 0; r < 8; ++r) {
            int gr = tile * 16 + r + hi * 8;
            float v = acc[r];
            if (gr < N) {
                out_pre[(size_t)gr * CCH + col] = v;
                s += v;
                s2 += v * v;
            }
        }
        atomicAdd(&colsum[col], s);
        atomicAdd(&colsumsq[col], s2);
    }
}

// ----------------------------------------------------------------- BN finalize
__global__ void gine_bn_stats(float* colsum, float* colsumsq,
                              const float* __restrict__ gamma,
                              const float* __restrict__ beta, int N) {
    int c = threadIdx.x;
    if (c >= CCH) return;
    float mean  = colsum[c] / (float)N;
    float var   = colsumsq[c] / (float)N - mean * mean;
    float scale = gamma[c] * rsqrtf(var + 1e-5f);
    float shift = beta[c] - mean * scale;
    colsum[c]   = scale;
    colsumsq[c] = shift;
}

__global__ void gine_bn_apply(const float* __restrict__ out_pre,
                              const float* __restrict__ scale,
                              const float* __restrict__ shift,
                              float* __restrict__ out, int n) {
    int i = blockIdx.x * blockDim.x + threadIdx.x;
    if (i < n) {
        int c = i & (CCH - 1);
        out[i] = fmaxf(out_pre[i] * scale[c] + shift[c], 0.0f);
    }
}

// ------------------------------------------------------------------- launcher
extern "C" void kernel_launch(void* const* d_in, const int* in_sizes, int n_in,
                              void* d_out, int out_size, void* d_ws, size_t ws_size,
                              hipStream_t stream) {
    const float*     x     = (const float*)d_in[0];
    const long long* ei    = (const long long*)d_in[1];
    const float*     eattr = (const float*)d_in[2];
    const float*     ew    = (const float*)d_in[3];
    const float*     eb    = (const float*)d_in[4];
    const float*     w1    = (const float*)d_in[5];
    const float*     b1    = (const float*)d_in[6];
    const float*     w2    = (const float*)d_in[7];
    const float*     b2    = (const float*)d_in[8];
    const float*     rw    = (const float*)d_in[9];
    const float*     rb    = (const float*)d_in[10];
    const float*     eps   = (const float*)d_in[11];
    const float*     gamma = (const float*)d_in[12];
    const float*     beta  = (const float*)d_in[13];

    const int N = in_sizes[0] / CCH;
    const int E = in_sizes[2];

    char* ws = (char*)d_ws;
    size_t o = 0;
    float* sums = (float*)(ws + o);      o += (size_t)N * CCH * sizeof(float);
    float* cnt  = (float*)(ws + o);      o += (size_t)N * sizeof(float);
    float* colsum   = (float*)(ws + o);  o += CCH * sizeof(float);
    float* colsumsq = (float*)(ws + o);  o += CCH * sizeof(float);
    o = (o + 255) & ~(size_t)255;
    __bf16* blob = (__bf16*)(ws + o);    o += 3 * 16384 * sizeof(__bf16);
    o = (o + 255) & ~(size_t)255;
    float* out_pre = (float*)(ws + o);

    // 1) zero sums/cnt/stats (contiguous from ws base)
    int zc = N * CCH + N + 2 * CCH;
    gine_zero<<<(zc + 255) / 256, 256, 0, stream>>>((float*)ws, zc);

    // 2) pack bf16 weight blobs in B-fragment order
    gine_pack_w<<<(3 * 16384 + 255) / 256, 256, 0, stream>>>(w1, w2, rw, blob);

    // 3) edge scatter-mean accumulation
    long long ethreads = (long long)E * 32;
    gine_edge<<<(unsigned)((ethreads + 255) / 256), 256, 0, stream>>>(
        x, ei, eattr, ew, eb, sums, cnt, E);

    // 4) fused node MLP with WMMA (one wave per 16-row tile)
    gine_node<<<(N + 15) / 16, 32, 0, stream>>>(
        x, sums, cnt, blob, b1, b2, rb, eps, out_pre, colsum, colsumsq, N);

    // 5) finalize BN stats
    gine_bn_stats<<<1, CCH, 0, stream>>>(colsum, colsumsq, gamma, beta, N);

    // 6) normalize + ReLU
    int n = N * CCH;
    gine_bn_apply<<<(n + 255) / 256, 256, 0, stream>>>(
        out_pre, colsum, colsumsq, (float*)d_out, n);
}
